// AttentionBranch_34376918237826
// MI455X (gfx1250) — matrix operand order repaired and checked by reading, not compiled, not run on hardware
//
#include <hip/hip_runtime.h>
#include <hip/hip_bf16.h>

// ---------------------------------------------------------------------------
// B=4, Cin=2048, Cmid=512, NC=19, H=W=64 -> M = 16384 pixels.
// Heavy GEMMs (v_wmma_f32_16x16x32_bf16, f32 accumulate):
//   conva: M x 512, K = 9*2048 ; convb: M x 512, K = 512 ; b1: M x 512, K = 9*2560
// ~700 GFLOP vs ~0.5 GB traffic -> compute bound -> bf16 WMMA.
// Tiles stream global->LDS via GLOBAL_LOAD_ASYNC_TO_LDS_B128 (ASYNCcnt):
// no VGPR staging registers, so nothing can spill across the K-loop barrier.
// Zero-padded 66x66 NHWC activations make all 9 taps unconditional loads.
// ---------------------------------------------------------------------------

typedef __attribute__((ext_vector_type(16))) __bf16 v16bf;
typedef __attribute__((ext_vector_type(8)))  float  v8f;
typedef int v4i __attribute__((vector_size(16)));   // builtin's expected elt type
typedef __attribute__((address_space(3))) v4i* lds_v4i_ptr;

#define MTOT    16384
#define OCH     512
#define XCATC   2560
#define PADW    66
#define PADP    (66 * 66)
#define ASTRIDE 40             // padded LDS row stride (ushorts)

#if defined(__has_builtin)
#if __has_builtin(__builtin_amdgcn_global_load_async_to_lds_b128)
#define USE_ASYNC_LDS 1
#endif
#endif
#ifndef USE_ASYNC_LDS
#define USE_ASYNC_LDS 0
#endif

__device__ __forceinline__ void wait_async_zero()
{
#if defined(__has_builtin)
#if __has_builtin(__builtin_amdgcn_s_wait_asynccnt)
    __builtin_amdgcn_s_wait_asynccnt(0);
    return;
#else
    asm volatile("s_wait_asynccnt 0x0" ::: "memory");
    return;
#endif
#else
    asm volatile("s_wait_asynccnt 0x0" ::: "memory");
#endif
}

// ------- pack x (fp32 NCHW) -> xpad (bf16 padded-NHWC, ch 0..2047) ----------
__global__ __launch_bounds__(256)
void pack_x_kernel(const float* __restrict__ x, __hip_bfloat16* __restrict__ xpad)
{
    __shared__ float tile[32][33];
    const int tx = threadIdx.x, ty = threadIdx.y;     // block (32,8)
    const int hw0 = blockIdx.x * 32;
    const int c0  = blockIdx.y * 32;
    const int b   = blockIdx.z;
#pragma unroll
    for (int j = 0; j < 4; ++j) {
        int c = c0 + ty + j * 8;
        tile[ty + j * 8][tx] = x[(((size_t)b * 2048 + c) << 12) + hw0 + tx];
    }
    __syncthreads();
#pragma unroll
    for (int j = 0; j < 4; ++j) {
        int pl = ty + j * 8;
        int hwp = hw0 + pl;
        int h = hwp >> 6, w = hwp & 63;
        size_t pp = (size_t)b * PADP + (h + 1) * PADW + (w + 1);
        xpad[pp * XCATC + c0 + tx] = __float2bfloat16(tile[tx][pl]);
    }
}

// -------- pack weights: src fp32 [O][C][taps] -> dst bf16 [tap][O][C] -------
__global__ __launch_bounds__(256)
void pack_w_kernel(const float* __restrict__ src, __hip_bfloat16* __restrict__ dst,
                   int O, int C, int taps)
{
    int idx = blockIdx.x * 256 + threadIdx.x;
    int total = O * C * taps;
    if (idx >= total) return;
    int tap = idx / (O * C);
    int rem = idx - tap * O * C;
    int o = rem / C;
    int c = rem - o * C;
    dst[idx] = __float2bfloat16(src[((size_t)o * C + c) * taps + tap]);
}

// ---------------------------------------------------------------------------
// WMMA implicit-GEMM conv. Block tile 128(M) x 128(N) x 32(K).
// 8 waves = 4(M) x 2(N); wave tile 32 x 64 = 2x4 WMMA 16x16 tiles.
// Double-buffered LDS, one barrier per K-step. Streams advance 32-bit element
// offsets; tap boundaries apply precomputed deltas (no divisions). Prefetch is
// unconditional (ws guard slack absorbs the one dead tile past the end).
// ---------------------------------------------------------------------------
template<int TAPS>
__global__ __launch_bounds__(256)
__attribute__((amdgpu_waves_per_eu(2)))
void wmma_conv_gemm(const __hip_bfloat16* __restrict__ in, int rowStride, int C,
                    const __hip_bfloat16* __restrict__ wgt,
                    float* __restrict__ outF,
                    __hip_bfloat16* __restrict__ outB, int outStride, int outOfs,
                    int padOut, const float* __restrict__ bias)
{
    __shared__ __align__(16) unsigned short A_s[2][128 * ASTRIDE];
    __shared__ __align__(16) unsigned short B_s[2][128 * ASTRIDE];

    const unsigned short* __restrict__ inU = (const unsigned short*)in;   // SGPR base
    const unsigned short* __restrict__ wgU = (const unsigned short*)wgt;  // SGPR base

    const int tid   = threadIdx.x;
    const int lane  = tid & 31;
    const int wave  = tid >> 5;
    const int waveM = wave >> 1;          // 0..3
    const int waveN = wave & 1;           // 0..1
    const int m0 = blockIdx.y * 128;
    const int n0 = blockIdx.x * 128;

    // ---- per-thread staging coordinates (A: 2 chunks, B: 2 chunks)
    const int ccA = (tid & 3) * 8;
    int arow[2], ab[2], ah[2], aw[2];
#pragma unroll
    for (int i = 0; i < 2; ++i) {
        int chunk = tid + i * 256;
        int row = chunk >> 2;             // 0..127
        arow[i] = row;
        int p = m0 + row;
        ab[i] = p >> 12;
        int hw = p & 4095;
        ah[i] = hw >> 6;
        aw[i] = hw & 63;
    }
    int bnrow[2], ccB[2];
#pragma unroll
    for (int i = 0; i < 2; ++i) {
        int chunk = tid + i * 256;
        bnrow[i] = chunk >> 2;
        ccB[i]   = (chunk & 3) * 8;
    }
    const int sA[2] = { arow[0] * ASTRIDE + ccA,      arow[1] * ASTRIDE + ccA };
    const int sB[2] = { bnrow[0] * ASTRIDE + ccB[0],  bnrow[1] * ASTRIDE + ccB[1] };

    // ---- fragment-read offsets (16-bit A 16x32 / B 32x16 VGPR layouts)
    const int laneM = lane & 15;
    const int koffA = (lane & 16) ? 8 : 0;
    const int koffB = (lane & 16) ? 8 : 0;
    int rA[2], rB[4];
#pragma unroll
    for (int wm = 0; wm < 2; ++wm)
        rA[wm] = (waveM * 32 + wm * 16 + laneM) * ASTRIDE + koffA;
#pragma unroll
    for (int wn = 0; wn < 4; ++wn)
        rB[wn] = (waveN * 64 + wn * 16 + laneM) * ASTRIDE + koffB;

    v8f zero;
#pragma unroll
    for (int k = 0; k < 8; ++k) zero[k] = 0.0f;
    v8f acc[2][4];
#pragma unroll
    for (int i = 0; i < 2; ++i)
#pragma unroll
        for (int j = 0; j < 4; ++j) acc[i][j] = zero;

    // ---- streaming cursor: 32-bit element offsets + division-free tap bumps
    const int kcPerTap = C >> 5;
    const int ksteps   = TAPS * kcPerTap;
    unsigned aoff[2], boff[2];
#pragma unroll
    for (int i = 0; i < 2; ++i) {
        unsigned pp;
        if (TAPS == 9)      // tap 0 => dy=dx=-1 => padded coords (ah, aw)
            pp = (unsigned)(ab[i] * PADP + ah[i] * PADW + aw[i]);
        else
            pp = (unsigned)(m0 + arow[i]);
        aoff[i] = pp * (unsigned)rowStride + (unsigned)ccA;
        boff[i] = (unsigned)(n0 + bnrow[i]) * (unsigned)C + (unsigned)ccB[i];
    }
    const unsigned aDeltaN = (unsigned)(rowStride - C);        // dx step
    const unsigned aDeltaW = (unsigned)(64 * rowStride - C);   // dy step (wrap)
    const unsigned bDelta  = (unsigned)(OCH - 1) * (unsigned)C;
    int lkc = 0, dxc = 0;

    auto advance = [&]() {
        aoff[0] += 32; aoff[1] += 32;
        boff[0] += 32; boff[1] += 32;
        if (TAPS == 9) {
            if (++lkc == kcPerTap) {       // uniform scalar branch, adds only
                lkc = 0;
                unsigned ad = (dxc == 2) ? aDeltaW : aDeltaN;
                dxc = (dxc == 2) ? 0 : dxc + 1;
                aoff[0] += ad; aoff[1] += ad;
                boff[0] += bDelta; boff[1] += bDelta;
            }
        }
    };

#if USE_ASYNC_LDS
    // ---- async global->LDS staging: no VGPR staging, ASYNCcnt tracked ------
    auto issueTile = [&](int tb) {
#pragma unroll
        for (int i = 0; i < 2; ++i)
            __builtin_amdgcn_global_load_async_to_lds_b128(
                (v4i*)(inU + aoff[i]),
                (lds_v4i_ptr)&A_s[tb][sA[i]], 0, 0);
#pragma unroll
        for (int i = 0; i < 2; ++i)
            __builtin_amdgcn_global_load_async_to_lds_b128(
                (v4i*)(wgU + boff[i]),
                (lds_v4i_ptr)&B_s[tb][sB[i]], 0, 0);
        advance();
    };

    issueTile(0);
    wait_async_zero();
    __syncthreads();
    int buf = 0;
    for (int kt = 0; kt < ksteps; ++kt) {
        issueTile(buf ^ 1);                // tile kt+1 (dead on last iter; guard)
        const unsigned short* As = &A_s[buf][0];
        const unsigned short* Bs = &B_s[buf][0];
        union FB { uint4 u[2]; v16bf v; } fa[2], fb[4];
#pragma unroll
        for (int wm = 0; wm < 2; ++wm) {
            fa[wm].u[0] = *reinterpret_cast<const uint4*>(&As[rA[wm]]);
            fa[wm].u[1] = *reinterpret_cast<const uint4*>(&As[rA[wm] + 16]);
        }
#pragma unroll
        for (int wn = 0; wn < 4; ++wn) {
            fb[wn].u[0] = *reinterpret_cast<const uint4*>(&Bs[rB[wn]]);
            fb[wn].u[1] = *reinterpret_cast<const uint4*>(&Bs[rB[wn] + 16]);
        }
#pragma unroll
        for (int wn = 0; wn < 4; ++wn)
#pragma unroll
            for (int wm = 0; wm < 2; ++wm)
                acc[wm][wn] = __builtin_amdgcn_wmma_f32_16x16x32_bf16(
                    false, fa[wm].v, false, fb[wn].v, (short)0, acc[wm][wn],
                    false, false);
        wait_async_zero();                 // own async writes done
        __syncthreads();                   // everyone's writes visible
        buf ^= 1;
    }
#else
    // ---- fallback: register staging, unconditional prefetch ---------------
    uint4 avN[2], bvN[2];
    auto loadTile = [&]() {
        avN[0] = *reinterpret_cast<const uint4*>(inU + aoff[0]);
        avN[1] = *reinterpret_cast<const uint4*>(inU + aoff[1]);
        bvN[0] = *reinterpret_cast<const uint4*>(wgU + boff[0]);
        bvN[1] = *reinterpret_cast<const uint4*>(wgU + boff[1]);
        advance();
    };

    loadTile();
    int buf = 0;
    for (int kt = 0; kt < ksteps; ++kt) {
        unsigned short* As = &A_s[buf][0];
        unsigned short* Bs = &B_s[buf][0];
        *reinterpret_cast<uint4*>(&As[sA[0]]) = avN[0];
        *reinterpret_cast<uint4*>(&As[sA[1]]) = avN[1];
        *reinterpret_cast<uint4*>(&Bs[sB[0]]) = bvN[0];
        *reinterpret_cast<uint4*>(&Bs[sB[1]]) = bvN[1];
        __syncthreads();
        loadTile();                        // unconditional (guard slack in ws)

        union FB { uint4 u[2]; v16bf v; } fa[2], fb[4];
#pragma unroll
        for (int wm = 0; wm < 2; ++wm) {
            fa[wm].u[0] = *reinterpret_cast<const uint4*>(&As[rA[wm]]);
            fa[wm].u[1] = *reinterpret_cast<const uint4*>(&As[rA[wm] + 16]);
        }
#pragma unroll
        for (int wn = 0; wn < 4; ++wn) {
            fb[wn].u[0] = *reinterpret_cast<const uint4*>(&Bs[rB[wn]]);
            fb[wn].u[1] = *reinterpret_cast<const uint4*>(&Bs[rB[wn] + 16]);
        }
#pragma unroll
        for (int wn = 0; wn < 4; ++wn)
#pragma unroll
            for (int wm = 0; wm < 2; ++wm)
                acc[wm][wn] = __builtin_amdgcn_wmma_f32_16x16x32_bf16(
                    false, fa[wm].v, false, fb[wn].v, (short)0, acc[wm][wn],
                    false, false);
        buf ^= 1;
    }
#endif

    // ---- epilogue: C/D layout (lane0..15 -> N, lane bit4 -> +8 rows, vgpr -> row)
    const int rAdd = (lane & 16) ? 8 : 0;
#pragma unroll
    for (int wm = 0; wm < 2; ++wm)
#pragma unroll
        for (int wn = 0; wn < 4; ++wn) {
            int gc = n0 + waveN * 64 + wn * 16 + laneM;
            float bz = bias ? bias[gc] : 0.0f;
#pragma unroll
            for (int v = 0; v < 8; ++v) {
                int gr = m0 + waveM * 32 + wm * 16 + v + rAdd;
                float val = acc[wm][wn][v];
                if (outF) outF[(size_t)gr * OCH + gc] = val;
                if (outB) {
                    size_t orow;
                    if (padOut) {
                        int b = gr >> 12, hw = gr & 4095;
                        orow = (size_t)b * PADP + ((hw >> 6) + 1) * PADW
                               + ((hw & 63) + 1);
                    } else {
                        orow = (size_t)gr;
                    }
                    outB[orow * outStride + outOfs + gc] =
                        __float2bfloat16(val + bz);
                }
            }
        }
}

// -------- BN statistics: per-channel mean & rstd over 16384 rows ------------
__global__ __launch_bounds__(256)
void bn_stats_kernel(const float* __restrict__ raw,
                     float* __restrict__ mean, float* __restrict__ rstd)
{
    __shared__ float s1[256], s2[256];
    const int ch = blockIdx.x;
    const int tid = threadIdx.x;
    float a = 0.0f, b = 0.0f;
    for (int r = tid; r < MTOT; r += 256) {
        float v = raw[(size_t)r * OCH + ch];
        a += v; b += v * v;
    }
    s1[tid] = a; s2[tid] = b;
    __syncthreads();
    for (int s = 128; s > 0; s >>= 1) {
        if (tid < s) { s1[tid] += s1[tid + s]; s2[tid] += s2[tid + s]; }
        __syncthreads();
    }
    if (tid == 0) {
        float m = s1[0] * (1.0f / MTOT);
        float var = s2[0] * (1.0f / MTOT) - m * m;   // biased, as torch/jnp
        mean[ch] = m;
        rstd[ch] = rsqrtf(var + 1e-5f);
    }
}

// -------- BN + ReLU -> bf16 activations (stage A, dense [M][512]) -----------
__global__ __launch_bounds__(256)
void bn_relu_bf16_kernel(const float* __restrict__ raw,
                         const float* __restrict__ mean, const float* __restrict__ rstd,
                         const float* __restrict__ g, const float* __restrict__ bt,
                         __hip_bfloat16* __restrict__ outb)
{
    int idx = blockIdx.x * 256 + threadIdx.x;
    int ch = idx & (OCH - 1);
    float v = (raw[idx] - mean[ch]) * rstd[ch] * g[ch] + bt[ch];
    v = fmaxf(v, 0.0f);
    outb[idx] = __float2bfloat16(v);
}

// -------- BN + ReLU -> fp32 NCHW output AND bf16 activations (stage C) ------
__global__ __launch_bounds__(256)
void bn_relu_out_kernel(const float* __restrict__ raw,
                        const float* __restrict__ mean, const float* __restrict__ rstd,
                        const float* __restrict__ g, const float* __restrict__ bt,
                        float* __restrict__ outNCHW, __hip_bfloat16* __restrict__ outb)
{
    int idx = blockIdx.x * 256 + threadIdx.x;
    int ch = idx & (OCH - 1);
    int p  = idx >> 9;
    float v = (raw[idx] - mean[ch]) * rstd[ch] * g[ch] + bt[ch];
    v = fmaxf(v, 0.0f);
    outb[idx] = __float2bfloat16(v);
    int b = p >> 12, hw = p & 4095;
    outNCHW[(((size_t)b * OCH + ch) << 12) + hw] = v;
}

// -------- b4: 1x1 classifier (512 -> 19), fp32 NCHW output ------------------
__global__ __launch_bounds__(256)
void head_kernel(const __hip_bfloat16* __restrict__ act,
                 const float* __restrict__ W4, const float* __restrict__ b4,
                 float* __restrict__ out)
{
    int idx = blockIdx.x * 256 + threadIdx.x;
    if (idx >= MTOT * 19) return;
    int p = idx / 19;
    int n = idx - p * 19;
    const __hip_bfloat16* a = act + (size_t)p * OCH;
    const float* w = W4 + (size_t)n * OCH;
    float acc = b4[n];
#pragma unroll 8
    for (int c = 0; c < OCH; ++c)
        acc += __bfloat162float(a[c]) * w[c];
    int b = p >> 12, hw = p & 4095;
    out[(((size_t)b * 19 + n) << 12) + hw] = acc;
}

// ---------------------------------------------------------------------------
extern "C" void kernel_launch(void* const* d_in, const int* in_sizes, int n_in,
                              void* d_out, int out_size, void* d_ws, size_t ws_size,
                              hipStream_t stream)
{
    (void)in_sizes; (void)n_in; (void)out_size; (void)ws_size;

    const float* x   = (const float*)d_in[0];
    const float* Wa  = (const float*)d_in[1];
    const float* ga  = (const float*)d_in[2];
    const float* ba  = (const float*)d_in[3];
    const float* Wb  = (const float*)d_in[4];
    const float* bb  = (const float*)d_in[5];
    const float* W1  = (const float*)d_in[6];
    const float* g1  = (const float*)d_in[7];
    const float* b1  = (const float*)d_in[8];
    const float* W4  = (const float*)d_in[9];
    const float* b4v = (const float*)d_in[10];

    char* cur = (char*)d_ws;
    auto alloc = [&](size_t bytes) -> void* {
        void* p = (void*)cur;
        cur += (bytes + 4096 + 255) & ~(size_t)255;   // +4KB guard slack
        return p;
    };
    const size_t xpadBytes = (size_t)4 * PADP * XCATC * 2;
    __hip_bfloat16* xpad = (__hip_bfloat16*)alloc(xpadBytes);
    __hip_bfloat16* WaP  = (__hip_bfloat16*)alloc((size_t)9 * OCH * 2048 * 2);
    __hip_bfloat16* W1P  = (__hip_bfloat16*)alloc((size_t)9 * OCH * 2560 * 2);
    __hip_bfloat16* WbP  = (__hip_bfloat16*)alloc((size_t)OCH * OCH * 2);
    float*          rawA = (float*)alloc((size_t)MTOT * OCH * 4);
    __hip_bfloat16* actA = (__hip_bfloat16*)alloc((size_t)MTOT * OCH * 2);
    float* meanA = (float*)alloc(OCH * 4);
    float* rstdA = (float*)alloc(OCH * 4);
    float* meanC = (float*)alloc(OCH * 4);
    float* rstdC = (float*)alloc(OCH * 4);
    float*          rawC = rawA;   // safe alias (sequential stream order)
    __hip_bfloat16* actC = actA;   // safe alias

    // 0) zero padded activation image (halo must be 0 for both 3x3 convs)
    (void)hipMemsetAsync(xpad, 0, xpadBytes, stream);

    // 1) repack input & weights to bf16, GEMM-friendly layouts
    pack_x_kernel<<<dim3(128, 64, 4), dim3(32, 8), 0, stream>>>(x, xpad);
    pack_w_kernel<<<(9 * OCH * 2048 + 255) / 256, 256, 0, stream>>>(Wa, WaP, OCH, 2048, 9);
    pack_w_kernel<<<(9 * OCH * 2560 + 255) / 256, 256, 0, stream>>>(W1, W1P, OCH, 2560, 9);
    pack_w_kernel<<<(OCH * OCH + 255) / 256, 256, 0, stream>>>(Wb, WbP, OCH, OCH, 1);

    dim3 gGrid(OCH / 128, MTOT / 128), gBlk(256);

    // 2) conva (3x3, K=9*2048) -> rawA, BN stats + ReLU -> actA (bf16)
    wmma_conv_gemm<9><<<gGrid, gBlk, 0, stream>>>(xpad, XCATC, 2048, WaP,
                                                  rawA, nullptr, 0, 0, 0, nullptr);
    bn_stats_kernel<<<OCH, 256, 0, stream>>>(rawA, meanA, rstdA);
    bn_relu_bf16_kernel<<<(MTOT * OCH) / 256, 256, 0, stream>>>(rawA, meanA, rstdA,
                                                                ga, ba, actA);
    // 3) convb (1x1 + bias) -> padded xpad channels 2048..2559 (concat for free)
    wmma_conv_gemm<1><<<gGrid, gBlk, 0, stream>>>(actA, OCH, OCH, WbP,
                                                  nullptr, xpad, XCATC, 2048, 1, bb);
    // 4) b1 (3x3 over 2560-ch padded concat) -> rawC, BN stats
    wmma_conv_gemm<9><<<gGrid, gBlk, 0, stream>>>(xpad, XCATC, 2560, W1P,
                                                  rawC, nullptr, 0, 0, 0, nullptr);
    bn_stats_kernel<<<OCH, 256, 0, stream>>>(rawC, meanC, rstdC);

    // 5) BN+ReLU -> output #1 (fp32 NCHW) and bf16 activations for b4
    float* out0 = (float*)d_out;
    bn_relu_out_kernel<<<(MTOT * OCH) / 256, 256, 0, stream>>>(rawC, meanC, rstdC,
                                                               g1, b1, out0, actC);
    // 6) b4 classifier -> output #2
    head_kernel<<<(MTOT * 19 + 255) / 256, 256, 0, stream>>>(actC, W4, b4v,
                                                             out0 + (size_t)4 * OCH * 4096);
}
